// MutiHeadAttention_59356448031653
// MI455X (gfx1250) — compile-verified, matrix operand order checked by reading, not run on hardware
//
#include <hip/hip_runtime.h>
#include <hip/hip_bf16.h>
#include <math.h>

// ---------------------------------------------------------------------------
// GQA attention forward for MI455X (gfx1250, wave32, WMMA + async-to-LDS).
//   B=2, N=2048, D=1024, H=16, G=4, HD=64, hpg=4
// Pipeline (f16 staging in d_ws, f32 accumulation in WMMA):
//   1) convert x, Wq, Wk, Wv, Wo to f16
//   2) q/k/v projections: WMMA GEMM with double-buffered async-to-LDS tiles
//   3) flash attention per (b, head, 16-row q tile)  (WMMA + LDS)
//   4) out = y@Wo^T + bo (same GEMM, f32 out)
// ---------------------------------------------------------------------------

typedef __attribute__((ext_vector_type(16))) _Float16 v16h;
typedef __attribute__((ext_vector_type(8)))  _Float16 v8h;
typedef __attribute__((ext_vector_type(8)))  float    v8f;

#define ATT_B  2
#define ATT_N  2048
#define ATT_D  1024
#define ATT_G  4
#define ATT_HD 64
#define ATT_KD (ATT_G * ATT_HD)     // 256
#define ATT_M  (ATT_B * ATT_N)      // 4096 rows for the projections

// ---------------------------------------------------------------------------
// f32 -> f16 conversion (grid-stride)
// ---------------------------------------------------------------------------
__global__ void f32_to_f16_kernel(const float* __restrict__ src,
                                  _Float16* __restrict__ dst, int n) {
  int i = blockIdx.x * blockDim.x + threadIdx.x;
  int stride = gridDim.x * blockDim.x;
  for (; i < n; i += stride) dst[i] = (_Float16)src[i];
}

// Load one WMMA f16 fragment (ISA striping): halves 0..7 at p[0..7],
// halves 8..15 at p[16..23]; caller pre-offsets p by klo = (lane/16)*8.
__device__ inline v16h load_frag(const _Float16* p) {
  v8h lo = *(const v8h*)(p);
  v8h hi = *(const v8h*)(p + 16);
  v16h r;
#pragma unroll
  for (int i = 0; i < 8; ++i) { r[i] = lo[i]; r[i + 8] = hi[i]; }
  return r;
}

// ---------------------------------------------------------------------------
// Async stage one K-slab into LDS:
//   A rows [m0, m0+64) x cols [k0, k0+32)  -> As  (row-major [64][32] halves)
//   W rows [n0, n0+128) x cols [k0, k0+32) -> Ws  (row-major [128][32] halves)
// 128 threads; each issues 2 (A) + 4 (W) global_load_async_to_lds_b128,
// i.e. ASYNCcnt += 6 per wave per stage.
// ---------------------------------------------------------------------------
__device__ inline void stage_async(const _Float16* __restrict__ A,
                                   const _Float16* __restrict__ W,
                                   int K, int m0, int n0, int k0,
                                   _Float16* As, _Float16* Ws, int t) {
  const unsigned long long abase = (unsigned long long)(uintptr_t)A;
  const unsigned long long wbase = (unsigned long long)(uintptr_t)W;
#pragma unroll
  for (int i = 0; i < 2; ++i) {
    const int seg = t + i * 128;            // 0..255 (16B segments of A tile)
    const int row = seg >> 2;               // 0..63
    const int c8  = (seg & 3) * 8;          // half offset in row
    const unsigned int voff =
        (unsigned int)(((size_t)(m0 + row) * K + k0 + c8) * 2);
    const unsigned int lds = (unsigned int)(uintptr_t)(As + row * 32 + c8);
    asm volatile("global_load_async_to_lds_b128 %0, %1, %2"
                 :: "v"(lds), "v"(voff), "s"(abase) : "memory");
  }
#pragma unroll
  for (int i = 0; i < 4; ++i) {
    const int seg = t + i * 128;            // 0..511 (16B segments of W tile)
    const int row = seg >> 2;               // 0..127
    const int c8  = (seg & 3) * 8;
    const unsigned int voff =
        (unsigned int)(((size_t)(n0 + row) * K + k0 + c8) * 2);
    const unsigned int lds = (unsigned int)(uintptr_t)(Ws + row * 32 + c8);
    asm volatile("global_load_async_to_lds_b128 %0, %1, %2"
                 :: "v"(lds), "v"(voff), "s"(wbase) : "memory");
  }
}

// ---------------------------------------------------------------------------
// GEMM: out[m,n] = sum_k A[m,k] * W[n,k] + bias[n]
//   A: [M,K] f16 row-major, W: [N,K] f16 row-major, out: [M,N] OutT.
//   4 waves / block; block tile 64x128; wave tile 32x64; K-step 32.
//   Double-buffered async-to-LDS staging overlapped with WMMA.
// ---------------------------------------------------------------------------
template <typename OutT>
__global__ __launch_bounds__(128)
void gemm_wmma_kernel(const _Float16* __restrict__ A,
                      const _Float16* __restrict__ W,
                      const float* __restrict__ bias,
                      OutT* __restrict__ out,
                      int M, int N, int K) {
  __shared__ __align__(16) _Float16 AsBuf[2][64 * 32];   // 2 x 4 KB
  __shared__ __align__(16) _Float16 WsBuf[2][128 * 32];  // 2 x 8 KB

  const int t    = threadIdx.x;
  const int lane = t & 31;
  const int wv   = t >> 5;
  const int lm   = lane & 15;
  const int lh   = lane >> 4;
  const int klo  = lh * 8;
  const int m0   = blockIdx.x * 64;
  const int n0   = blockIdx.y * 128;
  const int mw   = (wv & 1) * 32;   // wave m-offset in block tile
  const int nw   = (wv >> 1) * 64;  // wave n-offset in block tile

  v8f acc[2][4];
#pragma unroll
  for (int mt = 0; mt < 2; ++mt)
#pragma unroll
    for (int nt = 0; nt < 4; ++nt) acc[mt][nt] = (v8f)(0.0f);

  const int nk = K / 32;
  // prologue: fill both buffers
  stage_async(A, W, K, m0, n0, 0, AsBuf[0], WsBuf[0], t);
  stage_async(A, W, K, m0, n0, 32, AsBuf[1], WsBuf[1], t);

  for (int i = 0; i < nk; ++i) {
    // wait for stage i (6 async per wave per stage; loads complete in order,
    // so <=6 outstanding means stage i has fully landed in LDS)
    if (i + 1 < nk) {
      asm volatile("s_wait_asynccnt 0x6" ::: "memory");
    } else {
      asm volatile("s_wait_asynccnt 0x0" ::: "memory");
    }
    __syncthreads();

    const _Float16* as = AsBuf[i & 1];
    const _Float16* ws = WsBuf[i & 1];
    v16h a[2], b[4];
#pragma unroll
    for (int mt = 0; mt < 2; ++mt)
      a[mt] = load_frag(as + (mw + mt * 16 + lm) * 32 + klo);
#pragma unroll
    for (int nt = 0; nt < 4; ++nt)
      b[nt] = load_frag(ws + (nw + nt * 16 + lm) * 32 + klo);
#pragma unroll
    for (int mt = 0; mt < 2; ++mt)
#pragma unroll
      for (int nt = 0; nt < 4; ++nt)
        acc[mt][nt] = __builtin_amdgcn_wmma_f32_16x16x32_f16(
            false, a[mt], false, b[nt], (short)0, acc[mt][nt], false, false);

    __syncthreads();  // everyone done reading buffer (i&1)
    if (i + 2 < nk)
      stage_async(A, W, K, m0, n0, (i + 2) * 32, AsBuf[i & 1], WsBuf[i & 1], t);
  }

  // C/D layout: lane, vgpr r -> (row = r + lh*8, col = lm) in each 16x16 tile
#pragma unroll
  for (int mt = 0; mt < 2; ++mt) {
#pragma unroll
    for (int nt = 0; nt < 4; ++nt) {
      const int col = n0 + nw + nt * 16 + lm;
      const float bb = bias[col];
#pragma unroll
      for (int r = 0; r < 8; ++r) {
        const int row = m0 + mw + mt * 16 + r + lh * 8;
        out[(size_t)row * N + col] = (OutT)(acc[mt][nt][r] + bb);
      }
    }
  }
}

// ---------------------------------------------------------------------------
// Flash attention: one wave per (b, head, 16-query tile).
//   q: [B,N,D] f16 (head hl at column hl*64), k/v: [B,N,KD] f16 (group g at
//   column g*64), y: [B,N,D] f16 output.
// ---------------------------------------------------------------------------
__global__ __launch_bounds__(32)
void attn_wmma_kernel(const _Float16* __restrict__ q,
                      const _Float16* __restrict__ k,
                      const _Float16* __restrict__ v,
                      _Float16* __restrict__ y) {
  __shared__ __align__(16) _Float16 vT[ATT_HD * 32];  // transposed V chunk [d][key]
  __shared__ __align__(16) _Float16 pT[16 * 32];      // P chunk, row-major [qrow][key]

  const int lane = threadIdx.x;
  const int lm   = lane & 15;
  const int lh   = lane >> 4;
  const int klo  = lh * 8;

  int idx = blockIdx.x;
  const int qt = idx & (ATT_N / 16 - 1);  idx >>= 7;   // N/16 = 128
  const int hl = idx & 15;                idx >>= 4;   // linear head 0..15
  const int b  = idx;                                   // batch
  const int g  = hl >> 2;                               // KV group
  const int q0 = qt * 16;

  // Q fragments: A-matrix 16x32 halves, d = 0..31 and 32..63
  const _Float16* qp =
      q + (size_t)(b * ATT_N + q0 + lm) * ATT_D + hl * ATT_HD + klo;
  const v16h aq0 = load_frag(qp);
  const v16h aq1 = load_frag(qp + 32);

  float mrun[8], lrun[8];
#pragma unroll
  for (int r = 0; r < 8; ++r) { mrun[r] = -INFINITY; lrun[r] = 0.0f; }
  v8f acc[4];
#pragma unroll
  for (int dc = 0; dc < 4; ++dc) acc[dc] = (v8f)(0.0f);

  const float scale = 0.125f;  // 1/sqrt(64)

  for (int kc = 0; kc < ATT_N; kc += 32) {
    // ---- scores: S = Q @ K^T for 32 keys (two 16x16 tiles, K=64 via 2 wmma)
    const _Float16* kp0 =
        k + (size_t)(b * ATT_N + kc + lm) * ATT_KD + g * ATT_HD + klo;
    const _Float16* kp1 = kp0 + (size_t)16 * ATT_KD;
    v8f s0 = (v8f)(0.0f), s1 = (v8f)(0.0f);
    s0 = __builtin_amdgcn_wmma_f32_16x16x32_f16(false, aq0, false,
        load_frag(kp0), (short)0, s0, false, false);
    s0 = __builtin_amdgcn_wmma_f32_16x16x32_f16(false, aq1, false,
        load_frag(kp0 + 32), (short)0, s0, false, false);
    s1 = __builtin_amdgcn_wmma_f32_16x16x32_f16(false, aq0, false,
        load_frag(kp1), (short)0, s1, false, false);
    s1 = __builtin_amdgcn_wmma_f32_16x16x32_f16(false, aq1, false,
        load_frag(kp1 + 32), (short)0, s1, false, false);

    // ---- stage V chunk transposed into LDS: vT[d][key], lane owns one key
    {
      const _Float16* vp =
          v + (size_t)(b * ATT_N + kc + lane) * ATT_KD + g * ATT_HD;
#pragma unroll
      for (int c = 0; c < 8; ++c) {
        v8h vv = *(const v8h*)(vp + c * 8);
#pragma unroll
        for (int j = 0; j < 8; ++j) vT[(c * 8 + j) * 32 + lane] = vv[j];
      }
    }

    // ---- online softmax over the 32 keys of this chunk
    float alpha[8], p0v[8], p1v[8];
#pragma unroll
    for (int r = 0; r < 8; ++r) {
      const float a0 = s0[r] * scale;
      const float a1 = s1[r] * scale;
      float tmx = fmaxf(a0, a1);
      tmx = fmaxf(tmx, __shfl_xor(tmx, 1, 32));
      tmx = fmaxf(tmx, __shfl_xor(tmx, 2, 32));
      tmx = fmaxf(tmx, __shfl_xor(tmx, 4, 32));
      tmx = fmaxf(tmx, __shfl_xor(tmx, 8, 32));
      const float mn = fmaxf(mrun[r], tmx);
      const float al = __expf(mrun[r] - mn);
      const float p0 = __expf(a0 - mn);
      const float p1 = __expf(a1 - mn);
      float rs = p0 + p1;
      rs += __shfl_xor(rs, 1, 32);
      rs += __shfl_xor(rs, 2, 32);
      rs += __shfl_xor(rs, 4, 32);
      rs += __shfl_xor(rs, 8, 32);
      lrun[r] = lrun[r] * al + rs;
      mrun[r] = mn;
      alpha[r] = al;
      p0v[r] = p0;
      p1v[r] = p1;
    }
#pragma unroll
    for (int dc = 0; dc < 4; ++dc)
#pragma unroll
      for (int r = 0; r < 8; ++r) acc[dc][r] *= alpha[r];

    // ---- write P to LDS (C layout -> row-major [16][32])
#pragma unroll
    for (int r = 0; r < 8; ++r) {
      const int row = r + lh * 8;
      pT[row * 32 + lm]      = (_Float16)p0v[r];
      pT[row * 32 + 16 + lm] = (_Float16)p1v[r];
    }
    __syncthreads();

    // ---- O += P @ V : A-frag of P (16x32), B-frags of V (32 x 16 per dc)
    const v16h pa = load_frag(pT + lm * 32 + klo);
#pragma unroll
    for (int dc = 0; dc < 4; ++dc) {
      const v16h bv = load_frag(vT + (dc * 16 + lm) * 32 + klo);
      acc[dc] = __builtin_amdgcn_wmma_f32_16x16x32_f16(
          false, pa, false, bv, (short)0, acc[dc], false, false);
    }
    __syncthreads();
  }

  // ---- epilogue: normalize and store y (f16)
#pragma unroll
  for (int r = 0; r < 8; ++r) {
    const float inv = 1.0f / lrun[r];
    const int row = q0 + r + lh * 8;
    _Float16* yp = y + (size_t)(b * ATT_N + row) * ATT_D + hl * ATT_HD;
#pragma unroll
    for (int dc = 0; dc < 4; ++dc)
      yp[dc * 16 + lm] = (_Float16)(acc[dc][r] * inv);
  }
}

// ---------------------------------------------------------------------------
// Host-side launch
// ---------------------------------------------------------------------------
extern "C" void kernel_launch(void* const* d_in, const int* in_sizes, int n_in,
                              void* d_out, int out_size, void* d_ws, size_t ws_size,
                              hipStream_t stream) {
  (void)in_sizes; (void)n_in; (void)out_size; (void)ws_size;

  const float* x  = (const float*)d_in[0];
  const float* Wq = (const float*)d_in[1];
  const float* bq = (const float*)d_in[2];
  const float* Wk = (const float*)d_in[3];
  const float* bk = (const float*)d_in[4];
  const float* Wv = (const float*)d_in[5];
  const float* bv = (const float*)d_in[6];
  const float* Wo = (const float*)d_in[7];
  const float* bo = (const float*)d_in[8];
  float* out = (float*)d_out;

  _Float16* ws = (_Float16*)d_ws;
  const size_t nX  = (size_t)ATT_M * ATT_D;   // 4194304
  const size_t nWq = (size_t)ATT_D * ATT_D;   // 1048576
  const size_t nWk = (size_t)ATT_KD * ATT_D;  // 262144
  _Float16* xh  = ws;
  _Float16* wqh = xh  + nX;
  _Float16* wkh = wqh + nWq;
  _Float16* wvh = wkh + nWk;
  _Float16* woh = wvh + nWk;
  _Float16* qh  = woh + nWq;
  _Float16* kh  = qh  + nX;
  _Float16* vh  = kh  + (size_t)ATT_M * ATT_KD;
  _Float16* yh  = vh  + (size_t)ATT_M * ATT_KD;

  // 1) f16 conversions
  f32_to_f16_kernel<<<2048, 256, 0, stream>>>(x,  xh,  (int)nX);
  f32_to_f16_kernel<<<1024, 256, 0, stream>>>(Wq, wqh, (int)nWq);
  f32_to_f16_kernel<<<256,  256, 0, stream>>>(Wk, wkh, (int)nWk);
  f32_to_f16_kernel<<<256,  256, 0, stream>>>(Wv, wvh, (int)nWk);
  f32_to_f16_kernel<<<1024, 256, 0, stream>>>(Wo, woh, (int)nWq);

  // 2) projections: 4-wave blocks, 64x128 tile
  dim3 blk(128, 1, 1);
  dim3 gq(ATT_M / 64, ATT_D / 128, 1);   // 64 x 8
  dim3 gkv(ATT_M / 64, ATT_KD / 128, 1); // 64 x 2
  gemm_wmma_kernel<_Float16><<<gq,  blk, 0, stream>>>(xh, wqh, bq, qh,
      ATT_M, ATT_D, ATT_D);
  gemm_wmma_kernel<_Float16><<<gkv, blk, 0, stream>>>(xh, wkh, bk, kh,
      ATT_M, ATT_KD, ATT_D);
  gemm_wmma_kernel<_Float16><<<gkv, blk, 0, stream>>>(xh, wvh, bv, vh,
      ATT_M, ATT_KD, ATT_D);

  // 3) attention: one wave per (b, head, 16-query tile) = 2*16*128 = 4096
  dim3 ablk(32, 1, 1);
  attn_wmma_kernel<<<ATT_B * 16 * (ATT_N / 16), ablk, 0, stream>>>(qh, kh, vh, yh);

  // 4) output projection, f32 straight into d_out
  gemm_wmma_kernel<float><<<gq, blk, 0, stream>>>(yh, woh, bo, out,
      ATT_M, ATT_D, ATT_D);
}
